// GCN_71116068488094
// MI455X (gfx1250) — matrix operand order repaired and checked by reading, not compiled, not run on hardware
//
#include <hip/hip_runtime.h>
#include <hip/hip_bf16.h>

// MI455X (gfx1250) GCN forward.
// Roofline: per layer the edge scatter moves ~1.7GB (dominant, ~75us @ 23.3TB/s);
// the GEMM is 3.28 GFLOP / 51MB -> memory bound, fed through
// v_wmma_f32_16x16x32_bf16 (wave32, 16x16 tiles, K=128 => 4 wmma per tile).
// Tail handling is wave-uniform (only the last block is partial) so the hot
// path has no per-lane EXEC masking around loads/stores.

typedef __attribute__((ext_vector_type(16))) __bf16 v16bf;
typedef __attribute__((ext_vector_type(8)))  __bf16 v8bf;
typedef __attribute__((ext_vector_type(8)))  float  v8f;

#define D_FEAT    128
#define WT_STRIDE 136   // padded bf16 stride (272B, 16B aligned) to spread LDS banks

__device__ __forceinline__ void atomic_fadd(float* p, float v) {
    __hip_atomic_fetch_add(p, v, __ATOMIC_RELAXED, __HIP_MEMORY_SCOPE_AGENT);
}

// ---------------- degree / dinv ----------------
__global__ void deg_init(float* deg, int n) {
    int i = blockIdx.x * blockDim.x + threadIdx.x;
    if (i < n) deg[i] = 1.0f;                   // self-loop
}
__global__ void deg_count(const int* __restrict__ dst, float* deg, int E) {
    int e = blockIdx.x * blockDim.x + threadIdx.x;
    if (e < E) atomic_fadd(&deg[dst[e]], 1.0f);
}
__global__ void deg_rsqrt(float* deg, int n) {
    int i = blockIdx.x * blockDim.x + threadIdx.x;
    if (i < n) deg[i] = rsqrtf(deg[i]);         // deg >= 1 always (self-loop)
}

// ---------------- WMMA GEMM: y[i,:] = dinv[i] * (Hin[i,:] @ W); acc = y ----------------
// Block: 128 threads = 4 waves; each wave computes a 16-row x 128-col tile.
__global__ __launch_bounds__(128)
void gcn_gemm_wmma(const float* __restrict__ Hin, const float* __restrict__ W,
                   const float* __restrict__ dinv, float* __restrict__ y,
                   float* __restrict__ acc, int n)
{
    __shared__ __bf16 Wt[D_FEAT * WT_STRIDE];   // W transposed: Wt[col][k], ~34KB LDS
    const int tid = threadIdx.x;
    // stage W (row-major [k][n]) transposed to bf16, float4 global reads
    for (int idx4 = tid * 4; idx4 < D_FEAT * D_FEAT; idx4 += 128 * 4) {
        const int k = idx4 >> 7;
        const int c = idx4 & 127;               // 4 consecutive columns, same k
        const float4 w4 = *(const float4*)(W + idx4);
        Wt[(c + 0) * WT_STRIDE + k] = (__bf16)w4.x;
        Wt[(c + 1) * WT_STRIDE + k] = (__bf16)w4.y;
        Wt[(c + 2) * WT_STRIDE + k] = (__bf16)w4.z;
        Wt[(c + 3) * WT_STRIDE + k] = (__bf16)w4.w;
    }
    __syncthreads();

    const int lane    = tid & 31;
    const int wave    = tid >> 5;
    const int hi      = lane >> 4;     // half-wave select
    const int l15     = lane & 15;
    const int rowbase = blockIdx.x * 64 + wave * 16;
    const bool full   = (rowbase + 16 <= n);         // wave-uniform fast path
    const int arow    = min(rowbase + l15, n - 1);   // clamped load row (always valid)

    // A fragments (16x32 bf16, row arow). ISA layout:
    // lanes 0-15:  v0..3 = K 0..7,  v4..7 = K 16..23
    // lanes 16-31: v0..3 = K 8..15, v4..7 = K 24..31
    v16bf a[4];
    {
        const float* hrow = Hin + (size_t)arow * D_FEAT;
        #pragma unroll
        for (int kt = 0; kt < 4; ++kt) {
            const int k0 = kt * 32 + hi * 8;
            const int k1 = k0 + 16;
            const float4 f0 = *(const float4*)(hrow + k0);
            const float4 f1 = *(const float4*)(hrow + k0 + 4);
            const float4 f2 = *(const float4*)(hrow + k1);
            const float4 f3 = *(const float4*)(hrow + k1 + 4);
            a[kt][0]  = (__bf16)f0.x; a[kt][1]  = (__bf16)f0.y;
            a[kt][2]  = (__bf16)f0.z; a[kt][3]  = (__bf16)f0.w;
            a[kt][4]  = (__bf16)f1.x; a[kt][5]  = (__bf16)f1.y;
            a[kt][6]  = (__bf16)f1.z; a[kt][7]  = (__bf16)f1.w;
            a[kt][8]  = (__bf16)f2.x; a[kt][9]  = (__bf16)f2.y;
            a[kt][10] = (__bf16)f2.z; a[kt][11] = (__bf16)f2.w;
            a[kt][12] = (__bf16)f3.x; a[kt][13] = (__bf16)f3.y;
            a[kt][14] = (__bf16)f3.z; a[kt][15] = (__bf16)f3.w;
        }
    }

    // per-lane output-row scales (C/D: lane holds column, VGPR r holds row r + 8*hi)
    // clamped load + select, no EXEC masking
    float dv[8];
    #pragma unroll
    for (int r = 0; r < 8; ++r) {
        const int m = rowbase + r + 8 * hi;
        const float d = dinv[min(m, n - 1)];
        dv[r] = (m < n) ? d : 0.0f;
    }

    #pragma unroll
    for (int nt = 0; nt < 8; ++nt) {
        const int ncol = nt * 16 + l15;
        const __bf16* wcol = &Wt[ncol * WT_STRIDE];

        // hoist all 4 K-step B fragments (32x16 each: lanes 0-15 hold K 0..7,16..23,
        // lanes 16-31 hold K 8..15,24..31) so the wmma chain isn't gated load-by-load
        v16bf b[4];
        #pragma unroll
        for (int kt = 0; kt < 4; ++kt) {
            const int kb = kt * 32 + hi * 8;
            const v8bf blo = *(const v8bf*)(wcol + kb);       // ds_load_b128
            const v8bf bhi = *(const v8bf*)(wcol + kb + 16);  // ds_load_b128
            b[kt] = __builtin_shufflevector(blo, bhi, 0, 1, 2, 3, 4, 5, 6, 7,
                                            8, 9, 10, 11, 12, 13, 14, 15);
        }

        v8f c = {};
        #pragma unroll
        for (int kt = 0; kt < 4; ++kt)
            c = __builtin_amdgcn_wmma_f32_16x16x32_bf16(
                    false, a[kt], false, b[kt], (short)0, c, false, false);

        float* yp = y   + (size_t)(rowbase + 8 * hi) * D_FEAT + ncol;
        float* ap = acc + (size_t)(rowbase + 8 * hi) * D_FEAT + ncol;
        if (full) {
            #pragma unroll
            for (int r = 0; r < 8; ++r) {
                const float v = c[r] * dv[r];
                yp[(size_t)r * D_FEAT] = v;
                ap[(size_t)r * D_FEAT] = v;
            }
        } else {
            #pragma unroll
            for (int r = 0; r < 8; ++r) {
                if (rowbase + r + 8 * hi < n) {
                    const float v = c[r] * dv[r];
                    yp[(size_t)r * D_FEAT] = v;
                    ap[(size_t)r * D_FEAT] = v;
                }
            }
        }
    }
}

// ---------------- edge scatter: acc[dst,:] += y[src,:]  (one wave per edge) ----------------
__global__ __launch_bounds__(256)
void gcn_scatter(const int* __restrict__ src, const int* __restrict__ dst,
                 const float* __restrict__ y, float* __restrict__ acc, int E)
{
    const int wid  = (int)((blockIdx.x * (size_t)blockDim.x + threadIdx.x) >> 5);
    const int lane = threadIdx.x & 31;
    if (wid >= E) return;
    const int s = src[wid];
    const int d = dst[wid];
    const float4 v = *(const float4*)(y + (size_t)s * D_FEAT + lane * 4);
    float* p = acc + (size_t)d * D_FEAT + lane * 4;
    atomic_fadd(p + 0, v.x);
    atomic_fadd(p + 1, v.y);
    atomic_fadd(p + 2, v.z);
    atomic_fadd(p + 3, v.w);
}

// ---------------- finalize: h = relu(dinv[node]*acc + bias), float4 ----------------
__global__ void gcn_finalize(const float* __restrict__ acc, const float* __restrict__ dinv,
                             const float* __restrict__ bias, float* __restrict__ hout,
                             int n)
{
    const size_t q = (size_t)blockIdx.x * blockDim.x + threadIdx.x; // quad index
    if (q >= (size_t)n * (D_FEAT / 4)) return;
    const int   node = (int)(q >> 5);          // 32 quads per node
    const int   col4 = (int)(q & 31) * 4;
    const float dnv  = dinv[node];
    const float4 av  = *(const float4*)(acc + (size_t)node * D_FEAT + col4);
    const float4 bv  = *(const float4*)(bias + col4);
    float4 o;
    o.x = fmaxf(dnv * av.x + bv.x, 0.0f);
    o.y = fmaxf(dnv * av.y + bv.y, 0.0f);
    o.z = fmaxf(dnv * av.z + bv.z, 0.0f);
    o.w = fmaxf(dnv * av.w + bv.w, 0.0f);
    *(float4*)(hout + (size_t)node * D_FEAT + col4) = o;
}

// ---------------- pooling (batch is sorted => run-length segmented sum) ----------------
__global__ void pool_zero(float* gsum, float* gcnt, int G_) {
    int i = blockIdx.x * blockDim.x + threadIdx.x;
    if (i < G_ * D_FEAT) gsum[i] = 0.0f;
    if (i < G_)          gcnt[i] = 0.0f;
}

#define POOL_ROWS 512
__global__ __launch_bounds__(128)
void gcn_pool(const float* __restrict__ h, const int* __restrict__ batch,
              float* __restrict__ gsum, float* __restrict__ gcnt, int n)
{
    const int tid   = threadIdx.x;           // column 0..127
    const int start = blockIdx.x * POOL_ROWS;
    const int end   = min(start + POOL_ROWS, n);
    float run = 0.0f;
    int gcur = -1;
    for (int i = start; i < end; ++i) {
        const int g = batch[i];
        if (g != gcur) {
            if (gcur >= 0) atomic_fadd(&gsum[gcur * D_FEAT + tid], run);
            run = 0.0f; gcur = g;
        }
        run += h[(size_t)i * D_FEAT + tid];
    }
    if (gcur >= 0) atomic_fadd(&gsum[gcur * D_FEAT + tid], run);
    if (tid == 0) {
        float crun = 0.0f; int gc = -1;
        for (int i = start; i < end; ++i) {
            const int g = batch[i];
            if (g != gc) { if (gc >= 0) atomic_fadd(&gcnt[gc], crun); crun = 0.0f; gc = g; }
            crun += 1.0f;
        }
        if (gc >= 0) atomic_fadd(&gcnt[gc], crun);
    }
}

// ---------------- head: pooled @ Wf + bf, softmax (G=16, C=16) ----------------
__global__ __launch_bounds__(256)
void gcn_head(const float* __restrict__ gsum, const float* __restrict__ gcnt,
              const float* __restrict__ Wf, const float* __restrict__ bfv,
              float* __restrict__ out, int G_)
{
    __shared__ float lg[16][17];
    const int tid = threadIdx.x;
    const int g = tid >> 4, c = tid & 15;
    const bool act = (g < G_);
    if (act) {
        const float ic = 1.0f / fmaxf(gcnt[g], 1.0f);
        float aacc = bfv[c];
        for (int d = 0; d < D_FEAT; ++d)
            aacc += gsum[g * D_FEAT + d] * ic * Wf[d * 16 + c];
        lg[g][c] = aacc;
    }
    __syncthreads();
    if (act) {
        float m = -1e30f;
        for (int j = 0; j < 16; ++j) m = fmaxf(m, lg[g][j]);
        float s = 0.0f;
        for (int j = 0; j < 16; ++j) s += __expf(lg[g][j] - m);
        out[g * 16 + c] = __expf(lg[g][c] - m) / s;
    }
}

extern "C" void kernel_launch(void* const* d_in, const int* in_sizes, int n_in,
                              void* d_out, int out_size, void* d_ws, size_t ws_size,
                              hipStream_t stream)
{
    const float* x   = (const float*)d_in[0];
    const int*   ei  = (const int*)  d_in[1];   // [2,E] flat
    const int*   bat = (const int*)  d_in[2];
    const float* W1  = (const float*)d_in[3];
    const float* b1  = (const float*)d_in[4];
    const float* W2  = (const float*)d_in[5];
    const float* b2  = (const float*)d_in[6];
    const float* W3  = (const float*)d_in[7];
    const float* b3  = (const float*)d_in[8];
    const float* Wf  = (const float*)d_in[9];
    const float* bfv = (const float*)d_in[10];
    float* out = (float*)d_out;

    const int N = in_sizes[0] / D_FEAT;
    const int E = in_sizes[1] / 2;
    const int G = out_size / 16;
    const int* srcI = ei;
    const int* dstI = ei + E;

    // workspace carve-up (floats): dinv[N] | y[N*128] | acc[N*128] | h[N*128] | gsum | gcnt
    float* ws   = (float*)d_ws;
    size_t off  = 0;
    float* dinv = ws + off; off += ((size_t)N + 63) & ~(size_t)63;
    float* y    = ws + off; off += (size_t)N * D_FEAT;
    float* acc  = ws + off; off += (size_t)N * D_FEAT;
    float* h    = ws + off; off += (size_t)N * D_FEAT;
    float* gsum = ws + off; off += (size_t)G * D_FEAT;
    float* gcnt = ws + off;

    // degrees -> dinv
    deg_init <<<(N + 255) / 256, 256, 0, stream>>>(dinv, N);
    deg_count<<<(E + 255) / 256, 256, 0, stream>>>(dstI, dinv, E);
    deg_rsqrt<<<(N + 255) / 256, 256, 0, stream>>>(dinv, N);

    const int gemm_blocks = (N + 63) / 64;
    const int scat_blocks = (E + 7) / 8;                  // 8 waves/block, 1 edge/wave
    const int fin_blocks  = (int)(((size_t)N * (D_FEAT / 4) + 255) / 256);

    // layer 1
    gcn_gemm_wmma<<<gemm_blocks, 128, 0, stream>>>(x, W1, dinv, y, acc, N);
    gcn_scatter  <<<scat_blocks, 256, 0, stream>>>(srcI, dstI, y, acc, E);
    gcn_finalize <<<fin_blocks, 256, 0, stream>>>(acc, dinv, b1, h, N);
    // layer 2
    gcn_gemm_wmma<<<gemm_blocks, 128, 0, stream>>>(h, W2, dinv, y, acc, N);
    gcn_scatter  <<<scat_blocks, 256, 0, stream>>>(srcI, dstI, y, acc, E);
    gcn_finalize <<<fin_blocks, 256, 0, stream>>>(acc, dinv, b2, h, N);
    // layer 3
    gcn_gemm_wmma<<<gemm_blocks, 128, 0, stream>>>(h, W3, dinv, y, acc, N);
    gcn_scatter  <<<scat_blocks, 256, 0, stream>>>(srcI, dstI, y, acc, E);
    gcn_finalize <<<fin_blocks, 256, 0, stream>>>(acc, dinv, b3, h, N);

    // pool + head
    pool_zero<<<(G * D_FEAT + 255) / 256, 256, 0, stream>>>(gsum, gcnt, G);
    gcn_pool <<<(N + POOL_ROWS - 1) / POOL_ROWS, 128, 0, stream>>>(h, bat, gsum, gcnt, N);
    gcn_head <<<1, 256, 0, stream>>>(gsum, gcnt, Wf, bfv, out, G);
}
